// SVDLinear_41558103556850
// MI455X (gfx1250) — 8-bit matrix operand order repaired and checked by reading, compiled, not run on hardware
//
#include <hip/hip_runtime.h>
#include <cstdint>

typedef __attribute__((ext_vector_type(8))) int   v8i;
typedef __attribute__((ext_vector_type(8))) float v8f;
typedef __attribute__((ext_vector_type(2))) float v2f;

#define QGROUP 64
#define QMAXF  7.0f
#define BM 128
#define BN 128
#define LDST 80

__global__ __launch_bounds__(256) void qgroup_kernel(
    const float* __restrict__ src, const float* __restrict__ smooth,
    int8_t* __restrict__ qdst, float* __restrict__ sdst,
    int rows, int K)
{
  const int gpr  = K >> 6;
  const int wave = blockIdx.x * (blockDim.x >> 5) + (threadIdx.x >> 5);
  const int lane = threadIdx.x & 31;
  if (wave >= rows * gpr) return;
  const int row = wave / gpr;
  const int g   = wave - row * gpr;
  const size_t base = (size_t)row * K + (size_t)g * QGROUP;

  float2 v = *(const float2*)(src + base + 2 * lane);
  if (smooth) {
    const float2 s = *(const float2*)(smooth + g * QGROUP + 2 * lane);
    v.x *= s.x; v.y *= s.y;
  }
  float m = fmaxf(fabsf(v.x), fabsf(v.y));
#pragma unroll
  for (int off = 16; off > 0; off >>= 1)
    m = fmaxf(m, __shfl_xor(m, off, 32));
  const float scale = fmaxf(m / QMAXF, 1e-8f);
  const float inv   = 1.0f / scale;
  int qa = (int)rintf(v.x * inv); qa = qa < -7 ? -7 : (qa > 7 ? 7 : qa);
  int qb = (int)rintf(v.y * inv); qb = qb < -7 ? -7 : (qb > 7 ? 7 : qb);
  const uint16_t packed = (uint16_t)((qa & 0xff) | ((qb & 0xff) << 8));
  *(uint16_t*)(qdst + base + 2 * lane) = packed;
  if (lane == 0) sdst[(size_t)row * gpr + g] = scale;
}

__global__ __launch_bounds__(256) void lora_down_kernel(
    const float* __restrict__ x, const float* __restrict__ smooth,
    const float* __restrict__ ld, float* __restrict__ t,
    int B, int K, int R)
{
  const int row  = blockIdx.x * (blockDim.x >> 5) + (threadIdx.x >> 5);
  const int lane = threadIdx.x & 31;
  if (row >= B || lane >= R) return;
  const float* xr = x + (size_t)row * K;
  float acc = 0.0f;
  for (int k = 0; k < K; k += 4) {
#pragma unroll
    for (int u = 0; u < 4; ++u) {
      const float xs = xr[k + u] * smooth[k + u];
      acc = fmaf(xs, ld[(size_t)(k + u) * R + lane], acc);
    }
  }
  t[(size_t)row * R + lane] = acc;
}

__global__ __launch_bounds__(256) void w4a4_svd_gemm_kernel(
    const int8_t* __restrict__ qx, const float* __restrict__ sx,
    const int8_t* __restrict__ qw, const float* __restrict__ sw,
    const float* __restrict__ t,   const float* __restrict__ up,
    const float* __restrict__ bias, float* __restrict__ out,
    int B, int K, int N, int R)
{
  const int ngroups = K >> 6;
  __shared__ __align__(16) unsigned char lds[BM * 32 * 4 + BN * 32 * 4];
  unsigned char* sA  = lds;
  unsigned char* sB  = lds + BM * LDST;
  float*         sSX = (float*)(lds + 2 * BM * LDST);
  float*         sSW = sSX + BM;

  const int tid   = threadIdx.x;
  const int lane  = tid & 31;
  const int wid   = tid >> 5;
  const int waveM = wid & 3;
  const int waveN = wid >> 2;
  const int half  = lane >> 4;
  const int l16   = lane & 15;
  const int blockM = blockIdx.y * BM;
  const int blockN = blockIdx.x * BN;

  const v8i izero = {};
  const v8f fzero = {};
  v8f acc[2][4];
#pragma unroll
  for (int i = 0; i < 2; ++i)
#pragma unroll
    for (int j = 0; j < 4; ++j) acc[i][j] = fzero;

  uint4 ra[2], rb[2];
  float rs;
  const int st_row = tid >> 2;
  const int st_c   = (tid & 3) * 16;

  auto load_tiles = [&](int g) {
#pragma unroll
    for (int j = 0; j < 2; ++j) {
      const int row = st_row + j * 64;
      ra[j] = *(const uint4*)(qx + (size_t)(blockM + row) * K +
                              (size_t)g * QGROUP + st_c);
      rb[j] = *(const uint4*)(qw + (size_t)(blockN + row) * K +
                              (size_t)g * QGROUP + st_c);
    }
    rs = (tid < BM) ? sx[(size_t)(blockM + tid) * ngroups + g]
                    : sw[(size_t)(blockN + (tid - BM)) * ngroups + g];
  };

  load_tiles(0);

  for (int g = 0; g < ngroups; ++g) {
    __syncthreads();
#pragma unroll
    for (int j = 0; j < 2; ++j) {
      const int row = st_row + j * 64;
      *(uint4*)(sA + row * LDST + st_c) = ra[j];
      *(uint4*)(sB + row * LDST + st_c) = rb[j];
    }
    if (tid < BM) sSX[tid] = rs; else sSW[tid - BM] = rs;
    __syncthreads();

    if (g + 1 < ngroups) load_tiles(g + 1);

    v8i afrag[2];
#pragma unroll
    for (int ms = 0; ms < 2; ++ms) {
      const unsigned char* ap =
          sA + (waveM * 32 + ms * 16 + l16) * LDST + half * 16;
#pragma unroll
      for (int p = 0; p < 4; ++p) {
        const int2 d = *(const int2*)(ap + (p >> 1) * 32 + (p & 1) * 8);
        afrag[ms][2 * p]     = d.x;
        afrag[ms][2 * p + 1] = d.y;
      }
    }
    v8i bfrag[4];
#pragma unroll
    for (int ns = 0; ns < 4; ++ns) {
      const unsigned char* bp =
          sB + (waveN * 64 + ns * 16 + l16) * LDST + half * 16;
      const int4 lo = *(const int4*)(bp);
      const int4 hi = *(const int4*)(bp + 32);
      bfrag[ns][0] = lo.x; bfrag[ns][1] = lo.y;
      bfrag[ns][2] = lo.z; bfrag[ns][3] = lo.w;
      bfrag[ns][4] = hi.x; bfrag[ns][5] = hi.y;
      bfrag[ns][6] = hi.z; bfrag[ns][7] = hi.w;
    }

    float sxv[2][8];
#pragma unroll
    for (int ms = 0; ms < 2; ++ms) {
      const float* sp = sSX + waveM * 32 + ms * 16 + half * 8;
      const float4 s0 = *(const float4*)(sp);
      const float4 s1 = *(const float4*)(sp + 4);
      sxv[ms][0] = s0.x; sxv[ms][1] = s0.y; sxv[ms][2] = s0.z; sxv[ms][3] = s0.w;
      sxv[ms][4] = s1.x; sxv[ms][5] = s1.y; sxv[ms][6] = s1.z; sxv[ms][7] = s1.w;
    }
    float swv[4];
#pragma unroll
    for (int ns = 0; ns < 4; ++ns)
      swv[ns] = sSW[waveN * 64 + ns * 16 + l16];

#pragma unroll
    for (int ms = 0; ms < 2; ++ms) {
#pragma unroll
      for (int ns = 0; ns < 4; ++ns) {
        v8i d = __builtin_amdgcn_wmma_i32_16x16x64_iu8(
             true, afrag[ms],  true, bfrag[ns],
            izero,  false,  false);
#pragma unroll
        for (int r = 0; r < 8; ++r)
          acc[ms][ns][r] += (float)d[r] * (sxv[ms][r] * swv[ns]);
      }
    }
  }

  __syncthreads();
  float* sT  = (float*)lds;
  float* sUp = (float*)(lds + BM * 32 * 4);
  for (int i = tid; i < BM * 32; i += 256) {
    const int row = i >> 5, c = i & 31;
    sT[i]  = t [(size_t)(blockM + row) * R + c];
    sUp[i] = up[(size_t)(blockN + row) * R + c];
  }
  __syncthreads();

#pragma unroll
  for (int ms = 0; ms < 2; ++ms) {
    const float* tp = sT + (waveM * 32 + ms * 16 + l16) * 32;
#pragma unroll
    for (int ns = 0; ns < 4; ++ns) {
      const float* upp = sUp + (waveN * 64 + ns * 16 + l16) * 32;
      v8f c = acc[ms][ns];
#pragma unroll
      for (int kk = 0; kk < 8; ++kk) {
        v2f av, bv;
        av.x = tp [kk * 4 + half * 2];
        av.y = tp [kk * 4 + half * 2 + 1];
        bv.x = upp[kk * 4 + half * 2];
        bv.y = upp[kk * 4 + half * 2 + 1];
        c = __builtin_amdgcn_wmma_f32_16x16x4_f32(
            false, av, false, bv, (short)0, c, false, false);
      }
      acc[ms][ns] = c;
    }
  }

#pragma unroll
  for (int ms = 0; ms < 2; ++ms) {
#pragma unroll
    for (int ns = 0; ns < 4; ++ns) {
      const int n = blockN + waveN * 64 + ns * 16 + l16;
      const float bz = bias[n];
#pragma unroll
      for (int r = 0; r < 8; ++r) {
        const int m = blockM + waveM * 32 + ms * 16 + half * 8 + r;
        out[(size_t)m * N + n] = acc[ms][ns][r] + bz;
      }
    }
  }
}

extern "C" void kernel_launch(void* const* d_in, const int* in_sizes, int n_in,
                              void* d_out, int out_size, void* d_ws, size_t ws_size,
                              hipStream_t stream) {
  const float* x      = (const float*)d_in[0];
  const float* wres   = (const float*)d_in[1];
  const float* ld     = (const float*)d_in[2];
  const float* up     = (const float*)d_in[3];
  const float* smooth = (const float*)d_in[4];
  const float* bias   = (const float*)d_in[5];
  float* out = (float*)d_out;

  const int K = in_sizes[4];
  const int N = in_sizes[5];
  const int B = in_sizes[0] / K;
  const int R = in_sizes[2] / K;
  const int ngroups = K / QGROUP;

  size_t off = 0;
  auto carve = [&](size_t bytes) -> void* {
    void* p = (char*)d_ws + off;
    off += (bytes + 255) & ~(size_t)255;
    return p;
  };
  int8_t* qx = (int8_t*)carve((size_t)B * K);
  int8_t* qw = (int8_t*)carve((size_t)N * K);
  float*  sx = (float*)carve((size_t)B * ngroups * sizeof(float));
  float*  sw = (float*)carve((size_t)N * ngroups * sizeof(float));
  float*  t  = (float*)carve((size_t)B * R * sizeof(float));

  {
    const int groups = B * ngroups;
    qgroup_kernel<<<(groups + 7) / 8, 256, 0, stream>>>(x, smooth, qx, sx, B, K);
  }
  {
    const int groups = N * ngroups;
    qgroup_kernel<<<(groups + 7) / 8, 256, 0, stream>>>(wres, nullptr, qw, sw, N, K);
  }
  lora_down_kernel<<<(B + 7) / 8, 256, 0, stream>>>(x, smooth, ld, t, B, K, R);
  dim3 grid(N / BN, B / BM);
  w4a4_svd_gemm_kernel<<<grid, 256, 0, stream>>>(qx, sx, qw, sw, t, up, bias, out,
                                                 B, K, N, R);
}
